// BlockSparseAttention_64278480552153
// MI455X (gfx1250) — compile-verified, matrix operand order checked by reading, not run on hardware
//
#include <hip/hip_runtime.h>

// ---------------- problem constants (B=1) ----------------
#define TT 4096
#define CCH 2048
#define HH 16
#define DD 128
#define QBLK 64
#define SCALE 0.08838834764831845f   // 1/sqrt(128)

typedef unsigned short u16;
typedef __attribute__((ext_vector_type(16))) __bf16        v16bf;
typedef __attribute__((ext_vector_type(16))) unsigned short v16u;
typedef __attribute__((ext_vector_type(8)))  float          v8f;
typedef __attribute__((ext_vector_type(4)))  unsigned int   v4u;
typedef int v4i_vs __attribute__((vector_size(16)));        // matches builtin param

union FragU { v16u v; v4u q[2]; u16 h[16]; };
union Chunk8 { v4u q; u16 h[8]; };

// ---- CDNA5 async global->LDS path (guarded; falls back to sync staging) ----
#if __has_builtin(__builtin_amdgcn_global_load_async_to_lds_b128) && \
    __has_builtin(__builtin_amdgcn_s_wait_asynccnt)
#define HAS_ASYNC 1
#else
#define HAS_ASYNC 0
#if defined(__AMDGCN__)
#warning "gfx1250 async-to-LDS builtins unavailable in this toolchain; using sync staging"
#endif
#endif

#if HAS_ASYNC
__device__ __forceinline__ void async_cp16(const void* g, void* l) {
  // signature (probe-revealed): first param is v4i* (typed, generic AS)
  __builtin_amdgcn_global_load_async_to_lds_b128((v4i_vs*)g, (v4i_vs*)l, 0, 0);
}
#endif

__device__ __forceinline__ u16 f2bf(float f) {
  unsigned x = __builtin_bit_cast(unsigned, f);
  x += 0x7FFFu + ((x >> 16) & 1u);          // round-to-nearest-even
  return (u16)(x >> 16);
}
__device__ __forceinline__ float bf2f(u16 h) {
  unsigned x = ((unsigned)h) << 16;
  return __builtin_bit_cast(float, x);
}
__device__ __forceinline__ v8f vzero8() {
  v8f z;
  #pragma unroll
  for (int i = 0; i < 8; ++i) z[i] = 0.0f;
  return z;
}

// Load one 16x32 bf16 A/B-style fragment from a row-major tile.
// lane 0..15 -> row (lane&15), K chunks {0..7, 16..23}
// lane 16..31 -> row (lane&15), K chunks {8..15, 24..31}
__device__ __forceinline__ v16bf frag_rm(const u16* base, int stride, int lane) {
  const u16* p = base + (lane & 15) * stride + ((lane & 16) ? 8 : 0);
  FragU f;
  f.q[0] = *(const v4u*)(p);
  f.q[1] = *(const v4u*)(p + 16);
  return __builtin_bit_cast(v16bf, f.v);
}

// ---------------- f32 -> bf16 conversion ----------------
__global__ __launch_bounds__(256) void cvt_f32_bf16(const float* __restrict__ s,
                                                    u16* __restrict__ d, int n) {
  int i = blockIdx.x * blockDim.x + threadIdx.x;
  if (i < n) d[i] = f2bf(s[i]);
}

// ---------------- GEMM: Y[M,N] = A[M,K] * W[N,K]^T ----------------
// grid = (M/128, N/128), block = 256 threads = 8 waves in a 4x2 grid.
// Wave (wm,wn) computes a 32x64 sub-tile = 2x4 WMMA tiles. K-step 32,
// double-buffered LDS, async global->LDS staging overlapped with compute.
__global__ __launch_bounds__(256)
void gemm_bf16_nt(const u16* __restrict__ A, const u16* __restrict__ W,
                  u16* __restrict__ outb, float* __restrict__ outf,
                  int M, int N, int K, int out_is_f32) {
  __shared__ u16 lA[2][128][40];  // 128 x 32, padded (16B-aligned chunks)
  __shared__ u16 lB[2][128][40];
  const int tid  = threadIdx.x;
  const int lane = tid & 31;
  const int w    = tid >> 5;   // 0..7
  const int wm   = w >> 1;     // 0..3 -> rows wm*32
  const int wn   = w & 1;      // 0..1 -> cols wn*64
  const int m0   = blockIdx.x * 128;
  const int n0   = blockIdx.y * 128;

  v8f acc[2][4];
  #pragma unroll
  for (int mt = 0; mt < 2; ++mt)
    #pragma unroll
    for (int nt = 0; nt < 4; ++nt) acc[mt][nt] = vzero8();

  // stage one 128x32 A slab + 128x32 B slab into buffer `buf`
  // (each thread handles 2 A-chunks + 2 B-chunks of 16B -> 4 async ops/wave)
  auto stage = [&](int buf, int k0) {
    #pragma unroll
    for (int j = 0; j < 2; ++j) {
      const int c  = tid + j * 256;        // 0..511
      const int r  = c >> 2;
      const int c8 = (c & 3) * 8;
      const u16* ga = &A[(size_t)(m0 + r) * K + k0 + c8];
      const u16* gb = &W[(size_t)(n0 + r) * K + k0 + c8];
#if HAS_ASYNC
      async_cp16(ga, &lA[buf][r][c8]);
      async_cp16(gb, &lB[buf][r][c8]);
#else
      *(v4u*)&lA[buf][r][c8] = *(const v4u*)ga;
      *(v4u*)&lB[buf][r][c8] = *(const v4u*)gb;
#endif
    }
  };

  stage(0, 0);
  int buf = 0;
  for (int k0 = 0; k0 < K; k0 += 32, buf ^= 1) {
    const bool has_next = (k0 + 32 < K);
    if (has_next) stage(buf ^ 1, k0 + 32);   // overlap next slab with compute
    if (k0 + 64 < K) {                        // warm L2 two slabs ahead
      __builtin_prefetch(&A[(size_t)(m0 + (tid >> 1)) * K + k0 + 64], 0, 1);
      __builtin_prefetch(&W[(size_t)(n0 + (tid >> 1)) * K + k0 + 64], 0, 1);
    }
#if HAS_ASYNC
    // async loads retire in order: waiting to <=4 drains the 4 ops that
    // filled the *current* buffer, leaving the just-issued 4 in flight.
    if (has_next) __builtin_amdgcn_s_wait_asynccnt(4);
    else          __builtin_amdgcn_s_wait_asynccnt(0);
#endif
    __syncthreads();

    v16bf af[2];
    #pragma unroll
    for (int mt = 0; mt < 2; ++mt)
      af[mt] = frag_rm(&lA[buf][wm * 32 + mt * 16][0], 40, lane);
    #pragma unroll
    for (int nt = 0; nt < 4; ++nt) {
      const v16bf bf = frag_rm(&lB[buf][wn * 64 + nt * 16][0], 40, lane);
      #pragma unroll
      for (int mt = 0; mt < 2; ++mt)
        acc[mt][nt] = __builtin_amdgcn_wmma_f32_16x16x32_bf16(
            false, af[mt], false, bf, (short)0, acc[mt][nt], false, false);
    }
    __syncthreads();  // protect buffer about to be overwritten next iteration
  }

  // C/D layout: vgpr i -> row (lane>=16 ? i+8 : i), col = lane&15
  const int rbase = (lane >> 4) * 8;
  const int col   = lane & 15;
  #pragma unroll
  for (int mt = 0; mt < 2; ++mt) {
    #pragma unroll
    for (int nt = 0; nt < 4; ++nt) {
      #pragma unroll
      for (int i = 0; i < 8; ++i) {
        const size_t m = (size_t)(m0 + wm * 32 + mt * 16 + rbase + i);
        const size_t n = (size_t)(n0 + wn * 64 + nt * 16 + col);
        if (out_is_f32) outf[m * N + n] = acc[mt][nt][i];
        else            outb[m * N + n] = f2bf(acc[mt][nt][i]);
      }
    }
  }
}

// ---------------- RoPE (in place, bf16 [T, H*D]) ----------------
__global__ __launch_bounds__(256) void rope_bf16(u16* __restrict__ qk) {
  int idx = blockIdx.x * blockDim.x + threadIdx.x;  // T*H*64 pairs
  if (idx >= TT * HH * 64) return;
  const int i = idx & 63;
  const int h = (idx >> 6) & (HH - 1);
  const int t = idx >> 10;
  const float freq = __powf(10000.0f, -(float)i / 64.0f);
  float s, c;
  __sincosf((float)t * freq, &s, &c);
  u16* p = qk + (size_t)t * CCH + h * DD;
  const float x1 = bf2f(p[i]);
  const float x2 = bf2f(p[i + 64]);
  p[i]      = f2bf(x1 * c - x2 * s);
  p[i + 64] = f2bf(x2 * c + x1 * s);
}

// ---------------- block-sparse flash attention ----------------
// grid = (T/64, H), block = 128 threads = 4 waves; wave w owns query rows
// qb*64 + w*16 .. +15. Per allowed key block: S=Q*K^T (WMMA), online softmax,
// P round-trips through LDS (C-layout -> A-layout), ctx += P*V (WMMA).
__global__ __launch_bounds__(128)
void attn_sparse(const u16* __restrict__ Q, const u16* __restrict__ Km,
                 const u16* __restrict__ Vm, u16* __restrict__ Ctx) {
  __shared__ u16 lK[64][136];    // K block, row-major [krow][d]
  __shared__ u16 lVT[128][72];   // V block transposed [d][krow]
  __shared__ u16 lP[4][16][72];  // per-wave P (16 x 64, padded)

  const int tid  = threadIdx.x;
  const int lane = tid & 31;
  const int w    = tid >> 5;
  const int qb   = blockIdx.x;
  const int h    = blockIdx.y;
  const int t0   = qb * 64 + w * 16;
  const int rbase = (lane >> 4) * 8;
  const int col   = lane & 15;

  // resident Q fragments (D=128 -> 4 chunks of 32)
  const u16* qbase = Q + (size_t)t0 * CCH + h * DD;
  v16bf qf[4];
  #pragma unroll
  for (int kc = 0; kc < 4; ++kc) qf[kc] = frag_rm(qbase + kc * 32, CCH, lane);

  v8f acc[8];
  #pragma unroll
  for (int i = 0; i < 8; ++i) acc[i] = vzero8();
  float mrow[8], lrow[8];
  #pragma unroll
  for (int i = 0; i < 8; ++i) { mrow[i] = -3.0e38f; lrow[i] = 0.0f; }

  for (int kb = 0; kb <= qb; ++kb) {
    // block mask: sliding window(16 blks) | global(2 blks) | dilated(every 4)
    const bool keep = ((qb - kb) <= 16) || (kb < 2) || ((kb & 3) == 0);
    if (!keep) continue;

    const u16* kg = Km + (size_t)(kb * 64) * CCH + h * DD;
    const u16* vg = Vm + (size_t)(kb * 64) * CCH + h * DD;

    // stage K row-major: 64x128 halves = 1024 16B-chunks (async DMA to LDS)
    #pragma unroll
    for (int i = tid; i < 1024; i += 128) {
      const int r = i >> 4, c8 = (i & 15) * 8;
#if HAS_ASYNC
      async_cp16(&kg[(size_t)r * CCH + c8], &lK[r][c8]);
#else
      *(v4u*)&lK[r][c8] = *(const v4u*)&kg[(size_t)r * CCH + c8];
#endif
    }
    // stage V transposed (data must pass through VGPRs anyway)
    #pragma unroll
    for (int i = tid; i < 1024; i += 128) {
      const int r = i >> 4, d0 = (i & 15) * 8;
      Chunk8 u; u.q = *(const v4u*)&vg[(size_t)r * CCH + d0];
      #pragma unroll
      for (int j = 0; j < 8; ++j) lVT[d0 + j][r] = u.h[j];
    }
#if HAS_ASYNC
    __builtin_amdgcn_s_wait_asynccnt(0);
#endif
    __syncthreads();

    // S = Q * K^T : 4 j-tiles x 4 k-chunks
    v8f sc[4];
    #pragma unroll
    for (int jt = 0; jt < 4; ++jt) {
      v8f c = vzero8();
      #pragma unroll
      for (int kc = 0; kc < 4; ++kc) {
        const v16bf kf = frag_rm(&lK[jt * 16][kc * 32], 136, lane);
        c = __builtin_amdgcn_wmma_f32_16x16x32_bf16(
            false, qf[kc], false, kf, (short)0, c, false, false);
      }
      sc[jt] = c;
    }

    // online softmax (row stats reduced across the 16-lane column groups)
    float mnew[8];
    #pragma unroll
    for (int i = 0; i < 8; ++i) {
      float r = fmaxf(fmaxf(sc[0][i], sc[1][i]), fmaxf(sc[2][i], sc[3][i]));
      r = fmaxf(r, __shfl_xor(r, 1));
      r = fmaxf(r, __shfl_xor(r, 2));
      r = fmaxf(r, __shfl_xor(r, 4));
      r = fmaxf(r, __shfl_xor(r, 8));
      mnew[i] = fmaxf(mrow[i], r * SCALE);
    }
    #pragma unroll
    for (int i = 0; i < 8; ++i) {
      const float alpha = __expf(mrow[i] - mnew[i]);
      lrow[i] *= alpha;
      #pragma unroll
      for (int dt = 0; dt < 8; ++dt) acc[dt][i] *= alpha;
      float psum = 0.0f;
      #pragma unroll
      for (int jt = 0; jt < 4; ++jt) {
        const float p = __expf(sc[jt][i] * SCALE - mnew[i]);
        psum += p;
        lP[w][rbase + i][jt * 16 + col] = f2bf(p);
      }
      psum += __shfl_xor(psum, 1);
      psum += __shfl_xor(psum, 2);
      psum += __shfl_xor(psum, 4);
      psum += __shfl_xor(psum, 8);
      lrow[i] += psum;
      mrow[i]  = mnew[i];
    }

    // ctx += P * V : 8 d-tiles x 2 k-chunks
    #pragma unroll
    for (int dt = 0; dt < 8; ++dt) {
      v8f c = acc[dt];
      #pragma unroll
      for (int kc = 0; kc < 2; ++kc) {
        const v16bf pf = frag_rm(&lP[w][0][kc * 32], 72, lane);
        const v16bf vf = frag_rm(&lVT[dt * 16][kc * 32], 72, lane);
        c = __builtin_amdgcn_wmma_f32_16x16x32_bf16(
            false, pf, false, vf, (short)0, c, false, false);
      }
      acc[dt] = c;
    }
    __syncthreads();
  }

  // normalize + store bf16 ctx [T, H*D]
  #pragma unroll
  for (int dt = 0; dt < 8; ++dt) {
    #pragma unroll
    for (int i = 0; i < 8; ++i) {
      const size_t t = (size_t)(t0 + rbase + i);
      Ctx[t * CCH + h * DD + dt * 16 + col] = f2bf(acc[dt][i] / lrow[i]);
    }
  }
}

// ---------------- host launcher ----------------
extern "C" void kernel_launch(void* const* d_in, const int* in_sizes, int n_in,
                              void* d_out, int out_size, void* d_ws, size_t ws_size,
                              hipStream_t stream) {
  const float* x  = (const float*)d_in[0];
  const float* wq = (const float*)d_in[1];
  const float* wk = (const float*)d_in[2];
  const float* wv = (const float*)d_in[3];
  const float* wo = (const float*)d_in[4];
  float* out = (float*)d_out;

  char* ws = (char*)d_ws;
  size_t off = 0;
  const size_t xbytes = (size_t)TT * CCH * sizeof(u16);
  const size_t wbytes = (size_t)CCH * CCH * sizeof(u16);
  u16* xb   = (u16*)(ws + off); off += xbytes;
  u16* wqb  = (u16*)(ws + off); off += wbytes;
  u16* wkb  = (u16*)(ws + off); off += wbytes;
  u16* wvb  = (u16*)(ws + off); off += wbytes;
  u16* wob  = (u16*)(ws + off); off += wbytes;
  u16* qbuf = (u16*)(ws + off); off += xbytes;
  u16* kbuf = (u16*)(ws + off); off += xbytes;
  u16* vbuf = (u16*)(ws + off); off += xbytes;
  u16* ctxb = (u16*)(ws + off); off += xbytes;

  const int nx = TT * CCH, nw = CCH * CCH;
  cvt_f32_bf16<<<(nx + 255) / 256, 256, 0, stream>>>(x,  xb,  nx);
  cvt_f32_bf16<<<(nw + 255) / 256, 256, 0, stream>>>(wq, wqb, nw);
  cvt_f32_bf16<<<(nw + 255) / 256, 256, 0, stream>>>(wk, wkb, nw);
  cvt_f32_bf16<<<(nw + 255) / 256, 256, 0, stream>>>(wv, wvb, nw);
  cvt_f32_bf16<<<(nw + 255) / 256, 256, 0, stream>>>(wo, wob, nw);

  dim3 gg(TT / 128, CCH / 128);  // (32, 16)
  gemm_bf16_nt<<<gg, 256, 0, stream>>>(xb, wqb, qbuf, nullptr, TT, CCH, CCH, 0);
  gemm_bf16_nt<<<gg, 256, 0, stream>>>(xb, wkb, kbuf, nullptr, TT, CCH, CCH, 0);
  gemm_bf16_nt<<<gg, 256, 0, stream>>>(xb, wvb, vbuf, nullptr, TT, CCH, CCH, 0);

  const int npairs = TT * HH * 64;
  rope_bf16<<<(npairs + 255) / 256, 256, 0, stream>>>(qbuf);
  rope_bf16<<<(npairs + 255) / 256, 256, 0, stream>>>(kbuf);

  dim3 ga(TT / QBLK, HH);  // (64, 16)
  attn_sparse<<<ga, 128, 0, stream>>>(qbuf, kbuf, vbuf, ctxb);

  gemm_bf16_nt<<<gg, 256, 0, stream>>>(ctxb, wob, nullptr, out, TT, CCH, CCH, 1);
}
